// VectorQuantiser_46016279609586
// MI455X (gfx1250) — matrix operand order repaired and checked by reading, not compiled, not run on hardware
//
#include <hip/hip_runtime.h>
#include <hip/hip_bf16.h>
#include <stdint.h>

typedef _Float16 f16;
typedef __attribute__((ext_vector_type(16))) _Float16 v16h;
typedef __attribute__((ext_vector_type(8)))  _Float16 v8h;
typedef __attribute__((ext_vector_type(8)))  float    v8f;
typedef __attribute__((ext_vector_type(4)))  int      v4i;

#define AS1 __attribute__((address_space(1)))
#define AS3 __attribute__((address_space(3)))

#if __has_builtin(__builtin_amdgcn_global_load_async_to_lds_b128) && \
    __has_builtin(__builtin_amdgcn_s_wait_asynccnt)
#define USE_ASYNC_LDS 1
#endif

// ---------------- problem sizes ----------------
#define NROWS 8192
#define DDIM  768
#define CDIM  256
#define PCODES 16384
#define PSLICES 4
#define PSLICE_LEN (PCODES / PSLICES)   // 4096

// ---------------- workspace layout (bytes, 256B aligned) ----------------
#define WS_ZQ    0u                                   // fp32 [N, c]   8 MB (pre-norm, normalized in place)
#define WS_ZQ16  (WS_ZQ + (size_t)NROWS*CDIM*4u)      // f16  [N, c]   4 MB
#define WS_CB16  (WS_ZQ16 + (size_t)NROWS*CDIM*2u)    // f16  [P, c]   8 MB (normalized)
#define WS_CO16  (WS_CB16 + (size_t)PCODES*CDIM*2u)   // f16  [N, c]   4 MB (codes)
#define WS_WINT  (WS_CO16 + (size_t)NROWS*CDIM*2u)    // f16  [c, d]   384 KB (W_in^T)
#define WS_WOUTT (WS_WINT + (size_t)CDIM*DDIM*2u)     // f16  [d, c]   384 KB (W_out^T)
#define WS_IDX   (WS_WOUTT + (size_t)DDIM*CDIM*2u)    // i32  [N]      32 KB
#define WS_Z16   (WS_IDX + (size_t)NROWS*4u)          // f16  [N, d]   12 MB
#define WS_PBV   (WS_Z16 + (size_t)NROWS*DDIM*2u)     // f32  [S, N]   128 KB partial best value
#define WS_PBI   (WS_PBV + (size_t)PSLICES*NROWS*4u)  // i32  [S, N]   128 KB partial best index

// ---------------- d_out layout (floats) ----------------
#define OUT_LOSS ((size_t)NROWS*DDIM)
#define OUT_IDX  (OUT_LOSS + (size_t)NROWS*CDIM)

// ---------------- WMMA helpers ----------------
__device__ __forceinline__ v8f wmma_acc(v16h a, v16h b, v8f c) {
  return __builtin_amdgcn_wmma_f32_16x16x32_f16(
      false, a, false, b, (short)0, c, false, false);
}

__device__ __forceinline__ v16h cat8(v8h lo, v8h hi) {
  return __builtin_shufflevector(lo, hi, 0,1,2,3,4,5,6,7,8,9,10,11,12,13,14,15);
}

// A-matrix 16x32 f16: lane m = lane&15; lanes 0-15 hold K kt..kt+15,
// lanes 16-31 hold K kt+16..kt+31.
__device__ __forceinline__ v16h load_a_f16(const f16* __restrict__ base, int ld, int m, int kt, int lane) {
  int hi8 = (lane & 16) ? 16 : 0;
  const f16* p = base + (size_t)m * ld + kt + hi8;
  v8h lo = *(const v8h*)p;
  v8h hi = *(const v8h*)(p + 8);
  return cat8(lo, hi);
}

// B-matrix 32x16 f16 from "B^T" storage: lane n = lane&15; lanes 0-15 hold
// K kt..kt+15, lanes 16-31 hold K kt+16..kt+31 (16 contiguous halves per lane).
__device__ __forceinline__ v16h load_bt_f16(const f16* __restrict__ bt, int ld, int n, int kt, int lane) {
  const f16* p = bt + (size_t)n * ld + kt + ((lane & 16) ? 16 : 0);
  return *(const v16h*)p;
}

__device__ __forceinline__ float block_reduce_sum_256(float v, float* sred) {
  int t = threadIdx.x;
  sred[t] = v; __syncthreads();
  #pragma unroll
  for (int s = 128; s > 0; s >>= 1) {
    if (t < s) sred[t] += sred[t + s];
    __syncthreads();
  }
  float r = sred[0];
  __syncthreads();
  return r;
}

// ---------------- kernels ----------------

// dst[C x R] = transpose(src[R x C]) -> f16
__global__ void k_transpose_f16(const float* __restrict__ src, f16* __restrict__ dst, int R, int C) {
  int i = blockIdx.x * blockDim.x + threadIdx.x;
  if (i >= R * C) return;
  int r = i / C, c = i % C;
  dst[(size_t)c * R + r] = (f16)src[(size_t)r * C + c];
}

// elementwise fp32 -> f16
__global__ void k_cvt_f16(const float* __restrict__ src, f16* __restrict__ dst, int n) {
  int i = blockIdx.x * blockDim.x + threadIdx.x;
  if (i < n) dst[i] = (f16)src[i];
}

// normalize codebook rows -> f16
__global__ void k_norm_cb(const float* __restrict__ cb, f16* __restrict__ cb16) {
  __shared__ float sred[256];
  int row = blockIdx.x, t = threadIdx.x;
  float v = cb[(size_t)row * CDIM + t];
  float ss = block_reduce_sum_256(v * v, sred);
  float o = v * rsqrtf(ss + 1e-12f);
  cb16[(size_t)row * CDIM + t] = (f16)o;
}

// zq_pre = z @ W_in  (A = z16, B = W_in^T f16)
__global__ void k_proj(const f16* __restrict__ z16, const f16* __restrict__ winT,
                       float* __restrict__ zq_pre) {
  int lane = threadIdx.x & 31, wave = threadIdx.x >> 5;
  int tile = blockIdx.x * 8 + wave;              // 512 * 16 tiles
  int mt = (tile >> 4) << 4;
  int nt = (tile & 15) << 4;
  int m = mt + (lane & 15);
  int n = nt + (lane & 15);
  v8f acc;
  #pragma unroll
  for (int i = 0; i < 8; ++i) acc[i] = 0.f;
  #pragma unroll
  for (int kt = 0; kt < DDIM; kt += 32) {
    v16h a = load_a_f16(z16, DDIM, m, kt, lane);
    v16h b = load_bt_f16(winT, DDIM, n, kt, lane);
    acc = wmma_acc(a, b, acc);
  }
  int rb = mt + ((lane & 16) ? 8 : 0);
  #pragma unroll
  for (int r = 0; r < 8; ++r)
    zq_pre[(size_t)(rb + r) * CDIM + n] = acc[r];
}

// zq = normalize(zq_pre + b_in) in place; also f16 copy
__global__ void k_norm_zq(float* __restrict__ zq, const float* __restrict__ b_in,
                          f16* __restrict__ zq16) {
  __shared__ float sred[256];
  int row = blockIdx.x, t = threadIdx.x;
  float v = zq[(size_t)row * CDIM + t] + b_in[t];
  float ss = block_reduce_sum_256(v * v, sred);
  float o = v * rsqrtf(ss + 1e-12f);
  zq[(size_t)row * CDIM + t]   = o;
  zq16[(size_t)row * CDIM + t] = (f16)o;
}

// ---- argmax of zq . cb over a P-slice (== argmin distance; both normalized) ----
// Grid (32, 4): x = row-block (256 rows), y = P slice (4096 codes).
// 8 waves/block, each wave owns 32 rows (2 M-tiles) so every LDS B tile feeds
// two WMMAs. Codebook double-buffered in LDS with padded rows (264 halves:
// conflict-free 4-bank shift/row). Staging uses gfx1250 async global->LDS
// copies (ASYNCcnt) when available, else VGPR-staged copies.
#define PT 64
#define BSTR 264          // padded LDS row stride in halves (= 33 uint4)
__global__ void k_argmax(const f16* __restrict__ zq16, const f16* __restrict__ cb16,
                         float* __restrict__ pbv, int* __restrict__ pbi) {
  __shared__ f16 sB[2][PT * BSTR];               // 2 x 33 KB
  int t = threadIdx.x, lane = t & 31, wave = t >> 5;
  int mt0 = blockIdx.x * 256 + wave * 32;
  int m0 = mt0 + (lane & 15);
  int slice = blockIdx.y;
  int pbase = slice * PSLICE_LEN;

  // preload A: 32 rows x K=256, reused across 4096 codes
  v16h A0[8], A1[8];
  #pragma unroll
  for (int kt = 0; kt < 8; ++kt) {
    A0[kt] = load_a_f16(zq16, CDIM, m0,      kt * 32, lane);
    A1[kt] = load_a_f16(zq16, CDIM, m0 + 16, kt * 32, lane);
  }

  float best0[8], best1[8]; int bidx0[8], bidx1[8];
  #pragma unroll
  for (int r = 0; r < 8; ++r) {
    best0[r] = -3.0e38f; bidx0[r] = 0;
    best1[r] = -3.0e38f; bidx1[r] = 0;
  }

  const int NTILES = PSLICE_LEN / PT;            // 64

#ifdef USE_ASYNC_LDS
  // gfx1250 async path: global -> LDS with no VGPR data movement (ASYNCcnt).
  auto async_tile = [&](int p0, f16* buf) {
    const f16* g = cb16 + (size_t)p0 * CDIM;
    #pragma unroll
    for (int j = 0; j < 8; ++j) {
      int i = t + j * 256;                       // 16B chunk id within 32KB tile
      __builtin_amdgcn_global_load_async_to_lds_b128(
          (AS1 v4i*)(g + (size_t)i * 8),
          (AS3 v4i*)((char*)buf + ((size_t)(i >> 5) * 33 + (i & 31)) * 16),
          0, 0);
    }
  };

  async_tile(pbase, (f16*)sB[0]);
  __builtin_amdgcn_s_wait_asynccnt(0);
  __syncthreads();

  for (int tile = 0; tile < NTILES; ++tile) {
    int p0 = pbase + tile * PT;
    const f16* cur = sB[tile & 1];
    if (tile + 1 < NTILES) async_tile(p0 + PT, (f16*)sB[(tile + 1) & 1]);

    #pragma unroll
    for (int sub = 0; sub < PT / 16; ++sub) {
      int prow = sub * 16 + (lane & 15);
      const f16* bp = cur + (size_t)prow * BSTR + ((lane & 16) ? 16 : 0);
      v8f acc0, acc1;
      #pragma unroll
      for (int i = 0; i < 8; ++i) { acc0[i] = 0.f; acc1[i] = 0.f; }
      #pragma unroll
      for (int kt = 0; kt < 8; ++kt) {
        v8h lo = *(const v8h*)(bp + kt * 32);
        v8h hi = *(const v8h*)(bp + kt * 32 + 8);
        v16h b = cat8(lo, hi);
        acc0 = wmma_acc(A0[kt], b, acc0);
        acc1 = wmma_acc(A1[kt], b, acc1);
      }
      int p = p0 + sub * 16 + (lane & 15);
      #pragma unroll
      for (int r = 0; r < 8; ++r) {
        if (acc0[r] > best0[r] || (acc0[r] == best0[r] && p < bidx0[r])) { best0[r] = acc0[r]; bidx0[r] = p; }
        if (acc1[r] > best1[r] || (acc1[r] == best1[r] && p < bidx1[r])) { best1[r] = acc1[r]; bidx1[r] = p; }
      }
    }

    __builtin_amdgcn_s_wait_asynccnt(0);         // my async loads into other buffer landed
    __syncthreads();                             // everyone done reading cur + loads visible
  }
#else
  // fallback: VGPR-staged double buffer
  uint4 stage[8];
  auto ld_tile = [&](int p0) {
    const uint4* g = (const uint4*)(cb16 + (size_t)p0 * CDIM);
    #pragma unroll
    for (int j = 0; j < 8; ++j) stage[j] = g[t + j * 256];
  };
  auto st_tile = [&](f16* buf) {
    uint4* s4 = (uint4*)buf;
    #pragma unroll
    for (int j = 0; j < 8; ++j) {
      int i = t + j * 256;
      s4[(i >> 5) * 33 + (i & 31)] = stage[j];
    }
  };

  ld_tile(pbase);
  st_tile((f16*)sB[0]);
  __syncthreads();

  for (int tile = 0; tile < NTILES; ++tile) {
    int p0 = pbase + tile * PT;
    const f16* cur = sB[tile & 1];
    if (tile + 1 < NTILES) ld_tile(p0 + PT);

    #pragma unroll
    for (int sub = 0; sub < PT / 16; ++sub) {
      int prow = sub * 16 + (lane & 15);
      const f16* bp = cur + (size_t)prow * BSTR + ((lane & 16) ? 16 : 0);
      v8f acc0, acc1;
      #pragma unroll
      for (int i = 0; i < 8; ++i) { acc0[i] = 0.f; acc1[i] = 0.f; }
      #pragma unroll
      for (int kt = 0; kt < 8; ++kt) {
        v8h lo = *(const v8h*)(bp + kt * 32);
        v8h hi = *(const v8h*)(bp + kt * 32 + 8);
        v16h b = cat8(lo, hi);
        acc0 = wmma_acc(A0[kt], b, acc0);
        acc1 = wmma_acc(A1[kt], b, acc1);
      }
      int p = p0 + sub * 16 + (lane & 15);
      #pragma unroll
      for (int r = 0; r < 8; ++r) {
        if (acc0[r] > best0[r] || (acc0[r] == best0[r] && p < bidx0[r])) { best0[r] = acc0[r]; bidx0[r] = p; }
        if (acc1[r] > best1[r] || (acc1[r] == best1[r] && p < bidx1[r])) { best1[r] = acc1[r]; bidx1[r] = p; }
      }
    }

    __syncthreads();
    if (tile + 1 < NTILES) st_tile((f16*)sB[(tile + 1) & 1]);
    __syncthreads();
  }
#endif

  // cross-lane (16-lane half) max+argmax reduction
  #pragma unroll
  for (int off = 1; off < 16; off <<= 1) {
    #pragma unroll
    for (int r = 0; r < 8; ++r) {
      float ov = __shfl_xor(best0[r], off, 32);
      int   oi = __shfl_xor(bidx0[r], off, 32);
      if (ov > best0[r] || (ov == best0[r] && oi < bidx0[r])) { best0[r] = ov; bidx0[r] = oi; }
      float pv = __shfl_xor(best1[r], off, 32);
      int   qi = __shfl_xor(bidx1[r], off, 32);
      if (pv > best1[r] || (pv == best1[r] && qi < bidx1[r])) { best1[r] = pv; bidx1[r] = qi; }
    }
  }
  if ((lane & 15) == 0) {
    int rb = mt0 + ((lane & 16) ? 8 : 0);
    size_t so = (size_t)slice * NROWS;
    #pragma unroll
    for (int r = 0; r < 8; ++r) {
      pbv[so + rb + r]      = best0[r];  pbi[so + rb + r]      = bidx0[r];
      pbv[so + rb + 16 + r] = best1[r];  pbi[so + rb + 16 + r] = bidx1[r];
    }
  }
}

// reduce the 4 P-slice partials per row (lowest index wins ties)
__global__ void k_argreduce(const float* __restrict__ pbv, const int* __restrict__ pbi,
                            int* __restrict__ idx) {
  int r = blockIdx.x * 256 + threadIdx.x;
  if (r >= NROWS) return;
  float bv = pbv[r]; int bi = pbi[r];
  #pragma unroll
  for (int s = 1; s < PSLICES; ++s) {
    float v = pbv[(size_t)s * NROWS + r];
    int   i = pbi[(size_t)s * NROWS + r];
    if (v > bv || (v == bv && i < bi)) { bv = v; bi = i; }
  }
  idx[r] = bi;
}

// gather codebook row, normalize, emit codes f16, loss, idx-as-float
__global__ void k_gather(const int* __restrict__ idx, const float* __restrict__ codebook,
                         const float* __restrict__ zq, f16* __restrict__ codes16,
                         float* __restrict__ loss_out, float* __restrict__ idx_out) {
  __shared__ float sred[256];
  int row = blockIdx.x, t = threadIdx.x;
  int id = idx[row];
  float v = codebook[(size_t)id * CDIM + t];
  float ss = block_reduce_sum_256(v * v, sred);
  float o = v * rsqrtf(ss + 1e-12f);
  codes16[(size_t)row * CDIM + t] = (f16)o;
  float d = zq[(size_t)row * CDIM + t] - o;
  loss_out[(size_t)row * CDIM + t] = 1.25f * d * d;   // (BETA + 1) * diff^2
  if (t == 0) idx_out[row] = (float)id;
}

// out = codes @ W_out + b_out
__global__ void k_outproj(const f16* __restrict__ codes16, const f16* __restrict__ woutT,
                          const float* __restrict__ b_out, float* __restrict__ out) {
  int lane = threadIdx.x & 31, wave = threadIdx.x >> 5;
  int tile = blockIdx.x * 8 + wave;              // 512 * 48 tiles
  int mt = (tile / 48) * 16;
  int nt = (tile % 48) * 16;
  int m = mt + (lane & 15);
  int n = nt + (lane & 15);
  v8f acc;
  #pragma unroll
  for (int i = 0; i < 8; ++i) acc[i] = 0.f;
  #pragma unroll
  for (int kt = 0; kt < CDIM; kt += 32) {
    v16h a = load_a_f16(codes16, CDIM, m, kt, lane);
    v16h b = load_bt_f16(woutT, CDIM, n, kt, lane);
    acc = wmma_acc(a, b, acc);
  }
  float bias = b_out[n];
  int rb = mt + ((lane & 16) ? 8 : 0);
  #pragma unroll
  for (int r = 0; r < 8; ++r)
    out[(size_t)(rb + r) * DDIM + n] = acc[r] + bias;
}

// ---------------- launcher ----------------
extern "C" void kernel_launch(void* const* d_in, const int* in_sizes, int n_in,
                              void* d_out, int out_size, void* d_ws, size_t ws_size,
                              hipStream_t stream) {
  const float* z        = (const float*)d_in[0];   // [8192, 768]
  const float* W_in     = (const float*)d_in[1];   // [768, 256]
  const float* b_in     = (const float*)d_in[2];   // [256]
  const float* W_out    = (const float*)d_in[3];   // [256, 768]
  const float* b_out    = (const float*)d_in[4];   // [768]
  const float* codebook = (const float*)d_in[5];   // [16384, 256]

  char* ws = (char*)d_ws;
  float* zq      = (float*)(ws + WS_ZQ);
  f16*   zq16    = (f16*)  (ws + WS_ZQ16);
  f16*   cb16    = (f16*)  (ws + WS_CB16);
  f16*   codes16 = (f16*)  (ws + WS_CO16);
  f16*   winT    = (f16*)  (ws + WS_WINT);
  f16*   woutT   = (f16*)  (ws + WS_WOUTT);
  int*   idx     = (int*)  (ws + WS_IDX);
  f16*   z16     = (f16*)  (ws + WS_Z16);
  float* pbv     = (float*)(ws + WS_PBV);
  int*   pbi     = (int*)  (ws + WS_PBI);

  float* out      = (float*)d_out;
  float* loss_out = (float*)d_out + OUT_LOSS;
  float* idx_out  = (float*)d_out + OUT_IDX;

  // 1) weight transposes (f16) + z -> f16
  k_transpose_f16<<<(DDIM * CDIM + 255) / 256, 256, 0, stream>>>(W_in, winT, DDIM, CDIM);
  k_transpose_f16<<<(CDIM * DDIM + 255) / 256, 256, 0, stream>>>(W_out, woutT, CDIM, DDIM);
  k_cvt_f16<<<(NROWS * DDIM + 255) / 256, 256, 0, stream>>>(z, z16, NROWS * DDIM);

  // 2) normalize codebook -> f16
  k_norm_cb<<<PCODES, 256, 0, stream>>>(codebook, cb16);

  // 3) zq_pre = z @ W_in (WMMA)
  k_proj<<<(NROWS / 16) * (CDIM / 16) / 8, 256, 0, stream>>>(z16, winT, zq);

  // 4) zq = normalize(zq_pre + b_in); f16 copy
  k_norm_zq<<<NROWS, 256, 0, stream>>>(zq, b_in, zq16);

  // 5) nearest code: P-sliced LDS double-buffered WMMA argmax + reduce
  dim3 agrid(NROWS / 256, PSLICES);
  k_argmax<<<agrid, 256, 0, stream>>>(zq16, cb16, pbv, pbi);
  k_argreduce<<<NROWS / 256, 256, 0, stream>>>(pbv, pbi, idx);

  // 6) gather + normalize codes, loss, idx output
  k_gather<<<NROWS, 256, 0, stream>>>(idx, codebook, zq, codes16, loss_out, idx_out);

  // 7) out = codes @ W_out + b_out (WMMA)
  k_outproj<<<(NROWS / 16) * (DDIM / 16) / 8, 256, 0, stream>>>(codes16, woutT, b_out, out);
}
